// DeepseekMoE_TaskSpecificExperts_86165633893004
// MI455X (gfx1250) — matrix operand order repaired and checked by reading, not compiled, not run on hardware
//
#include <hip/hip_runtime.h>
#include <hip/hip_bf16.h>
#include <math.h>

typedef __bf16 bf16;
typedef __attribute__((ext_vector_type(8)))  bf16  v8bf;
typedef __attribute__((ext_vector_type(16))) bf16  v16bf;
typedef __attribute__((ext_vector_type(8)))  float v8f;

#define TDIM  4096   // B*S tokens
#define HDIM  1024
#define EDIM  8
#define KSEL  2
#define IDIM  512
#define ISDIM 1024

__device__ __forceinline__ float gelu_erf(float x) {
    return 0.5f * x * (1.0f + erff(x * 0.70710678118654752f));
}

// ---------------------------------------------------------------------------
// Router: logits = (x + task_emb[tid]) @ gate_w^T ; softmax ; top-2 ; norm.
// One wave per token. Also counts tokens per expert.
// ---------------------------------------------------------------------------
__global__ __launch_bounds__(256) void router_kernel(
    const float* __restrict__ X, const float* __restrict__ gate_w,
    const float* __restrict__ task_emb, const int* __restrict__ task_id,
    int* __restrict__ tk_idx, float* __restrict__ tk_w, int* __restrict__ cnt)
{
    const int wid = threadIdx.x >> 5, lane = threadIdx.x & 31;
    const int t = blockIdx.x * 8 + wid;
    if (t >= TDIM) return;
    const float* te = task_emb + (long)task_id[0] * HDIM;
    float acc[EDIM];
#pragma unroll
    for (int e = 0; e < EDIM; ++e) acc[e] = 0.f;
    for (int h = lane; h < HDIM; h += 32) {
        float xv = X[(long)t * HDIM + h] + te[h];
#pragma unroll
        for (int e = 0; e < EDIM; ++e) acc[e] += xv * gate_w[e * HDIM + h];
    }
#pragma unroll
    for (int e = 0; e < EDIM; ++e)
#pragma unroll
        for (int s = 16; s > 0; s >>= 1) acc[e] += __shfl_xor(acc[e], s, 32);
    if (lane == 0) {
        float mx = acc[0];
#pragma unroll
        for (int e = 1; e < EDIM; ++e) mx = fmaxf(mx, acc[e]);
        float sc[EDIM]; float sum = 0.f;
#pragma unroll
        for (int e = 0; e < EDIM; ++e) { sc[e] = __expf(acc[e] - mx); sum += sc[e]; }
#pragma unroll
        for (int e = 0; e < EDIM; ++e) sc[e] /= sum;
        int i1 = 0;
#pragma unroll
        for (int e = 1; e < EDIM; ++e) if (sc[e] > sc[i1]) i1 = e;
        int i2 = (i1 == 0) ? 1 : 0;
#pragma unroll
        for (int e = 0; e < EDIM; ++e) if (e != i1 && sc[e] > sc[i2]) i2 = e;
        float d = sc[i1] + sc[i2] + 1e-20f;
        tk_idx[t * 2 + 0] = i1;  tk_w[t * 2 + 0] = sc[i1] / d;
        tk_idx[t * 2 + 1] = i2;  tk_w[t * 2 + 1] = sc[i2] / d;
        atomicAdd(&cnt[i1], 1);
        atomicAdd(&cnt[i2], 1);
    }
}

__global__ void scan_kernel(const int* __restrict__ cnt, int* __restrict__ off) {
    if (threadIdx.x == 0 && blockIdx.x == 0) {
        int a = 0;
        for (int e = 0; e < EDIM; ++e) { off[e] = a; a += cnt[e]; }
    }
}

__global__ void fill_kernel(const int* __restrict__ tk_idx, const float* __restrict__ tk_w,
                            const int* __restrict__ off, int* __restrict__ cursor,
                            int* __restrict__ tok_sorted, float* __restrict__ w_sorted)
{
    int t = blockIdx.x * blockDim.x + threadIdx.x;
    if (t >= TDIM) return;
#pragma unroll
    for (int k = 0; k < KSEL; ++k) {
        int e = tk_idx[t * 2 + k];
        int slot = atomicAdd(&cursor[e], 1);
        int pos = off[e] + slot;
        tok_sorted[pos] = t;
        w_sorted[pos] = tk_w[t * 2 + k];
    }
}

// f32 -> bf16, 4 elements per thread
__global__ void cvt_kernel(const float* __restrict__ src, bf16* __restrict__ dst, long n4) {
    long i = ((long)blockIdx.x * blockDim.x + threadIdx.x) * 4;
    if (i >= n4 * 4) return;
    const float4 f = *(const float4*)&src[i];
    bf16 b0 = (bf16)f.x, b1 = (bf16)f.y, b2 = (bf16)f.z, b3 = (bf16)f.w;
    dst[i + 0] = b0; dst[i + 1] = b1; dst[i + 2] = b2; dst[i + 3] = b3;
}

// act = bf16( gelu(g) * u * rowWeight ), rowWeight index = i >> shift
__global__ void act_kernel(const float* __restrict__ g, const float* __restrict__ u,
                           const float* __restrict__ wrow, bf16* __restrict__ act,
                           int shift, long n)
{
    long i = (long)blockIdx.x * blockDim.x + threadIdx.x;
    if (i >= n) return;
    float w = wrow ? wrow[i >> shift] : 1.0f;
    act[i] = (bf16)(gelu_erf(g[i]) * u[i] * w);
}

// ---------------------------------------------------------------------------
// Generic bf16 WMMA GEMM: C[rows,N] = A[rows,K](bf16) * B[K,N](f32 -> bf16).
// 128x128 block tile, 8 waves, each wave 64x32 via 4x2 WMMA 16x16x32 frags.
// Double-buffered LDS; A tiles staged with CDNA5 async global->LDS DMA.
// Supports gathered A rows (row_ids), per-expert base/count, atomic scatter.
// ---------------------------------------------------------------------------
template <int SCATTER>
__global__ __launch_bounds__(256) void gemm_kernel(
    const bf16* __restrict__ A, const float* __restrict__ B, long b_stride,
    float* __restrict__ C,
    const int* __restrict__ row_ids,   // null -> identity rows
    const int* __restrict__ offs,      // null -> base 0
    const int* __restrict__ cnts,      // null -> rows_const
    int rows_const,
    const int* __restrict__ scat_tok,  // tokens per slot (SCATTER only)
    int K, int N)
{
    const int e = blockIdx.z;
    const int base  = offs ? offs[e] : 0;
    const int nrows = cnts ? cnts[e] : rows_const;
    const int m0 = blockIdx.x * 128;
    if (m0 >= nrows) return;
    const int n0 = blockIdx.y * 128;
    const float* Bm = B + (long)e * b_stride;

    __shared__ __align__(16) bf16 As[2][128][32];   // [buf][m][k]
    __shared__ __align__(16) bf16 Bs[2][128][32];   // [buf][n][k] (transposed)

    const int tid = threadIdx.x;
    const int lane = tid & 31, wid = tid >> 5;
    const int wm = wid >> 2, wn = wid & 3;          // 2 x 4 wave grid
    const int khalf = lane >> 4, l16 = lane & 15;

    // ---- hoisted A-gather addresses (loop-invariant) ----
    int mA[2], kqA[2];
    unsigned long long aga[2];
#pragma unroll
    for (int i = 0; i < 2; ++i) {
        int c = tid + i * 256;                       // 512 chunks of 8 bf16
        mA[i] = c >> 2; kqA[i] = (c & 3) * 8;
        int gm = m0 + mA[i];
        int gmc = gm < nrows ? gm : (nrows - 1);
        long ar = row_ids ? (long)row_ids[base + gmc] : (long)(base + gmc);
        aga[i] = (unsigned long long)(uintptr_t)(A + ar * (long)K + kqA[i]);
    }
    // ---- hoisted B staging coords ----
    const int bn = tid & 127;                        // output column in tile
    const int bkb = (tid >> 7) * 16;                 // 16 consecutive k values

    v8f acc[4][2];
#pragma unroll
    for (int i = 0; i < 4; ++i)
#pragma unroll
        for (int j = 0; j < 2; ++j)
#pragma unroll
            for (int v = 0; v < 8; ++v) acc[i][j][v] = 0.f;

    // ---- staging helpers ----
    auto stageA = [&](int buf, int k0) {
#pragma unroll
        for (int i = 0; i < 2; ++i) {
            unsigned lds = (unsigned)(uintptr_t)&As[buf][mA[i]][kqA[i]];
            unsigned long long ga = aga[i] + (unsigned long long)k0 * 2u;
            asm volatile("global_load_async_to_lds_b128 %0, %1, off"
                         :: "v"(lds), "v"(ga) : "memory");
        }
    };
    auto stageB = [&](int buf, int k0) {
        const float* src = Bm + (long)(k0 + bkb) * N + n0 + bn;
        float tmp[16];
#pragma unroll
        for (int j = 0; j < 16; ++j) tmp[j] = src[(long)j * N];
        v8bf p0, p1;
#pragma unroll
        for (int j = 0; j < 8; ++j) { p0[j] = (bf16)tmp[j]; p1[j] = (bf16)tmp[8 + j]; }
        *(v8bf*)&Bs[buf][bn][bkb + 0] = p0;
        *(v8bf*)&Bs[buf][bn][bkb + 8] = p1;
    };

    int cur = 0;
    stageA(0, 0);
    stageB(0, 0);

    for (int k0 = 0; k0 < K; k0 += 32) {
        asm volatile("s_wait_asynccnt 0x0" ::: "memory");
        __syncthreads();
        if (k0 + 32 < K) { stageA(1 ^ cur, k0 + 32); stageB(1 ^ cur, k0 + 32); }

        // ---- fragments per ISA layout ----
        v16bf a[4], b[2];
#pragma unroll
        for (int mi = 0; mi < 4; ++mi) {
            int m = wm * 64 + mi * 16 + l16;
            v8bf lo = *(const v8bf*)&As[cur][m][khalf * 8];
            v8bf hi = *(const v8bf*)&As[cur][m][16 + khalf * 8];
            a[mi] = __builtin_shufflevector(lo, hi, 0,1,2,3,4,5,6,7,8,9,10,11,12,13,14,15);
        }
#pragma unroll
        for (int ni = 0; ni < 2; ++ni) {
            int n = wn * 32 + ni * 16 + l16;
            v8bf lo = *(const v8bf*)&Bs[cur][n][khalf * 8];
            v8bf hi = *(const v8bf*)&Bs[cur][n][16 + khalf * 8];
            b[ni] = __builtin_shufflevector(lo, hi, 0,1,2,3,4,5,6,7,8,9,10,11,12,13,14,15);
        }
#pragma unroll
        for (int mi = 0; mi < 4; ++mi)
#pragma unroll
            for (int ni = 0; ni < 2; ++ni)
                acc[mi][ni] = __builtin_amdgcn_wmma_f32_16x16x32_bf16(
                    false, a[mi], false, b[ni], (short)0, acc[mi][ni], false, false);
        cur ^= 1;
    }

    // ---- epilogue ----
#pragma unroll
    for (int mi = 0; mi < 4; ++mi) {
#pragma unroll
        for (int ni = 0; ni < 2; ++ni) {
            int n = n0 + wn * 32 + ni * 16 + l16;
#pragma unroll
            for (int v = 0; v < 8; ++v) {
                int m = m0 + wm * 64 + mi * 16 + khalf * 8 + v;
                if (m < nrows) {
                    float val = acc[mi][ni][v];
                    if (SCATTER) {
                        int tok = scat_tok[base + m];
                        atomicAdd(&C[(long)tok * N + n], val);
                    } else {
                        C[(long)(base + m) * N + n] = val;
                    }
                }
            }
        }
    }
}

// out = a0*local + a1*shared, alpha = softmax(x @ Wc). One wave per token.
__global__ __launch_bounds__(256) void combine_kernel(
    const float* __restrict__ X, const float* __restrict__ Wc,
    const float* __restrict__ local_buf, const float* __restrict__ shared_buf,
    float* __restrict__ out)
{
    const int wid = threadIdx.x >> 5, lane = threadIdx.x & 31;
    const int t = blockIdx.x * 8 + wid;
    if (t >= TDIM) return;
    float d0 = 0.f, d1 = 0.f;
    for (int h = lane; h < HDIM; h += 32) {
        float xv = X[(long)t * HDIM + h];
        d0 += xv * Wc[h * 2 + 0];
        d1 += xv * Wc[h * 2 + 1];
    }
#pragma unroll
    for (int s = 16; s > 0; s >>= 1) { d0 += __shfl_xor(d0, s, 32); d1 += __shfl_xor(d1, s, 32); }
    float mx = fmaxf(d0, d1);
    float e0 = __expf(d0 - mx), e1 = __expf(d1 - mx);
    float a0 = e0 / (e0 + e1), a1 = e1 / (e0 + e1);
    for (int h = lane; h < HDIM; h += 32) {
        long idx = (long)t * HDIM + h;
        out[idx] = a0 * local_buf[idx] + a1 * shared_buf[idx];
    }
}

extern "C" void kernel_launch(void* const* d_in, const int* in_sizes, int n_in,
                              void* d_out, int out_size, void* d_ws, size_t ws_size,
                              hipStream_t stream)
{
    (void)in_sizes; (void)n_in; (void)out_size; (void)ws_size;
    const float* X        = (const float*)d_in[0];
    const int*   task_id  = (const int*)  d_in[1];
    const float* gate_w   = (const float*)d_in[2];
    const float* task_emb = (const float*)d_in[3];
    const float* We_gate  = (const float*)d_in[4];
    const float* We_up    = (const float*)d_in[5];
    const float* We_down  = (const float*)d_in[6];
    const float* Ws_gate  = (const float*)d_in[7];
    const float* Ws_up    = (const float*)d_in[8];
    const float* Ws_down  = (const float*)d_in[9];
    const float* Wc       = (const float*)d_in[10];
    float* out = (float*)d_out;

    char* w = (char*)d_ws;
    auto alloc = [&](size_t bytes) -> char* {
        char* p = w; w += (bytes + 255) & ~(size_t)255; return p;
    };
    bf16*  Xb         = (bf16*) alloc((size_t)TDIM * HDIM * 2);
    float* g_buf      = (float*)alloc((size_t)TDIM * KSEL * IDIM * 4);  // == T*IS f32
    float* u_buf      = (float*)alloc((size_t)TDIM * KSEL * IDIM * 4);
    bf16*  act_buf    = (bf16*) alloc((size_t)TDIM * KSEL * IDIM * 2);  // == T*IS bf16
    float* local_buf  = (float*)alloc((size_t)TDIM * HDIM * 4);
    float* shared_buf = (float*)alloc((size_t)TDIM * HDIM * 4);
    int*   tk_idx     = (int*)  alloc((size_t)TDIM * KSEL * 4);
    float* tk_w       = (float*)alloc((size_t)TDIM * KSEL * 4);
    int*   tok_sorted = (int*)  alloc((size_t)TDIM * KSEL * 4);
    float* w_sorted   = (float*)alloc((size_t)TDIM * KSEL * 4);
    int*   cnt        = (int*)  alloc(EDIM * 4);
    int*   off        = (int*)  alloc(EDIM * 4);
    int*   cursor     = (int*)  alloc(EDIM * 4);

    hipMemsetAsync(cnt, 0, EDIM * 4, stream);
    hipMemsetAsync(cursor, 0, EDIM * 4, stream);
    hipMemsetAsync(local_buf, 0, (size_t)TDIM * HDIM * 4, stream);

    { long n4 = (long)TDIM * HDIM / 4;
      cvt_kernel<<<(int)((n4 + 255) / 256), 256, 0, stream>>>(X, Xb, n4); }

    router_kernel<<<TDIM / 8, 256, 0, stream>>>(X, gate_w, task_emb, task_id,
                                                tk_idx, tk_w, cnt);
    scan_kernel<<<1, 32, 0, stream>>>(cnt, off);
    fill_kernel<<<(TDIM + 255) / 256, 256, 0, stream>>>(tk_idx, tk_w, off, cursor,
                                                        tok_sorted, w_sorted);

    dim3 blk(256);
    // routed experts: gate & up projections over gathered token rows
    dim3 g_gu(TDIM / 128, IDIM / 128, EDIM);
    gemm_kernel<0><<<g_gu, blk, 0, stream>>>(Xb, We_gate, (long)HDIM * IDIM, g_buf,
        tok_sorted, off, cnt, 0, nullptr, HDIM, IDIM);
    gemm_kernel<0><<<g_gu, blk, 0, stream>>>(Xb, We_up, (long)HDIM * IDIM, u_buf,
        tok_sorted, off, cnt, 0, nullptr, HDIM, IDIM);
    { long n = (long)TDIM * KSEL * IDIM;
      act_kernel<<<(int)((n + 255) / 256), 256, 0, stream>>>(g_buf, u_buf, w_sorted,
                                                             act_buf, 9 /*log2 IDIM*/, n); }
    // routed down projection, atomic scatter-combine into local_buf
    dim3 g_dn(TDIM / 128, HDIM / 128, EDIM);
    gemm_kernel<1><<<g_dn, blk, 0, stream>>>(act_buf, We_down, (long)IDIM * HDIM,
        local_buf, nullptr, off, cnt, 0, tok_sorted, IDIM, HDIM);

    // shared expert (reuses g_buf/u_buf/act_buf — same footprints)
    dim3 g_sh(TDIM / 128, ISDIM / 128, 1);
    gemm_kernel<0><<<g_sh, blk, 0, stream>>>(Xb, Ws_gate, 0, g_buf,
        nullptr, nullptr, nullptr, TDIM, nullptr, HDIM, ISDIM);
    gemm_kernel<0><<<g_sh, blk, 0, stream>>>(Xb, Ws_up, 0, u_buf,
        nullptr, nullptr, nullptr, TDIM, nullptr, HDIM, ISDIM);
    { long n = (long)TDIM * ISDIM;
      act_kernel<<<(int)((n + 255) / 256), 256, 0, stream>>>(g_buf, u_buf, nullptr,
                                                             act_buf, 10 /*log2 ISDIM*/, n); }
    dim3 g_sd(TDIM / 128, HDIM / 128, 1);
    gemm_kernel<0><<<g_sd, blk, 0, stream>>>(act_buf, Ws_down, 0, shared_buf,
        nullptr, nullptr, nullptr, TDIM, nullptr, ISDIM, HDIM);

    combine_kernel<<<TDIM / 8, 256, 0, stream>>>(X, Wc, local_buf, shared_buf, out);
}
